// ResMLPMoEBlock_84172769067206
// MI455X (gfx1250) — compile-verified, matrix operand order checked by reading, not run on hardware
//
#include <hip/hip_runtime.h>
#include <math.h>

#define Hdim 1024
#define NTOK 4096
#define NE   8
#define MLPH 4096
#define EXPH 2048
#define LDT  40   // padded LDS tile stride (elements): bank-conflict-free, 16B-aligned

typedef __attribute__((ext_vector_type(16))) __bf16 v16bf;
typedef __attribute__((ext_vector_type(8)))  __bf16 v8bf;
typedef __attribute__((ext_vector_type(8)))  float  v8f;

// ---------------- math helpers ----------------
__device__ __forceinline__ float gelu_f(float x) {
    // exact GELU (matches jax.nn.gelu approximate=False)
    return 0.5f * x * (1.0f + erff(x * 0.70710678118654752440f));
}

// ---------------- WMMA fragment load ----------------
// A and B tiles stored fragment-major in LDS: [major][LDT], major = M-row (A)
// or N-col (B). A lane's 16 bf16 values are two contiguous 16B runs ->
// two ds_load_b128. ISA K striping: lanes0-15 kb=0, lanes16-31 kb=8;
// values K {kb..kb+7, kb+16..kb+23}.
__device__ __forceinline__ v16bf frag_ld(const __bf16* base, int major, int lane) {
    const __bf16* p = base + major * LDT + ((lane < 16) ? 0 : 8);
    v8bf lo = *(const v8bf*)p;
    v8bf hi = *(const v8bf*)(p + 16);
    return __builtin_shufflevector(lo, hi, 0, 1, 2, 3, 4, 5, 6, 7,
                                   8, 9, 10, 11, 12, 13, 14, 15);
}
__device__ __forceinline__ v8f wmma_bf16(v16bf a, v16bf b, v8f c) {
    return __builtin_amdgcn_wmma_f32_16x16x32_bf16(false, a, false, b, (short)0, c, false, false);
}

// ---------------- LDS staging (256 threads) ----------------
// stage 32(K) x 64(N) f32 weight tile -> bf16 LDS, column-major [col][k].
__device__ __forceinline__ void stage_b(const float* __restrict__ gB, int ldb,
                                        __bf16* Bs, int tid) {
    int col = tid & 63;
    int k0  = (tid >> 6) * 8;   // 0,8,16,24
    __bf16 t[8];
#pragma unroll
    for (int i = 0; i < 8; ++i) t[i] = (__bf16)gB[(long)(k0 + i) * ldb + col];
    *(uint4*)(Bs + col * LDT + k0) = *(const uint4*)t;
}
// stage 128(M) x 32(K) bf16 A tile from contiguous rows (gA pre-offset to tile origin)
__device__ __forceinline__ void stage_a(const __bf16* __restrict__ gA, int lda_,
                                        __bf16* As, int tid) {
    int r = tid >> 1, c = (tid & 1) * 16;
    const __bf16* s = gA + (long)r * lda_ + c;
    *(uint4*)(As + r * LDT + c)     = *(const uint4*)s;
    *(uint4*)(As + r * LDT + c + 8) = *(const uint4*)(s + 8);
}
// stage 128x32 A tile with per-row gather through an index table
__device__ __forceinline__ void stage_a_rows(const __bf16* __restrict__ base, int lda_,
                                             const int* rows, int koff,
                                             __bf16* As, int tid) {
    int r = tid >> 1, c = (tid & 1) * 16;
    const __bf16* s = base + (long)rows[r] * lda_ + koff + c;
    *(uint4*)(As + r * LDT + c)     = *(const uint4*)s;
    *(uint4*)(As + r * LDT + c + 8) = *(const uint4*)(s + 8);
}

// ---------------- LayerNorm kernels ----------------
__global__ __launch_bounds__(256) void k_ln1(const float* __restrict__ x,
                                             const float* __restrict__ g,
                                             const float* __restrict__ b,
                                             __bf16* __restrict__ xh) {
    int row = blockIdx.x, tid = threadIdx.x;
    const float* xr = x + (long)row * Hdim;
    float v[4], s = 0.f, sq = 0.f;
#pragma unroll
    for (int j = 0; j < 4; ++j) { v[j] = xr[tid * 4 + j]; s += v[j]; sq += v[j] * v[j]; }
    __shared__ float s1[256], s2[256];
    s1[tid] = s; s2[tid] = sq; __syncthreads();
    for (int st = 128; st > 0; st >>= 1) {
        if (tid < st) { s1[tid] += s1[tid + st]; s2[tid] += s2[tid + st]; }
        __syncthreads();
    }
    float mu = s1[0] * (1.f / Hdim);
    float var = s2[0] * (1.f / Hdim) - mu * mu;
    float rs = rsqrtf(var + 1e-5f);
#pragma unroll
    for (int j = 0; j < 4; ++j) {
        int c = tid * 4 + j;
        xh[(long)row * Hdim + c] = (__bf16)(((v[j] - mu) * rs) * g[c] + b[c]);
    }
}

// LN2: x2 = LN(h)+h ; writes f32 and bf16 copies
__global__ __launch_bounds__(256) void k_ln2(const float* __restrict__ h,
                                             const float* __restrict__ g,
                                             const float* __restrict__ b,
                                             float* __restrict__ x2,
                                             __bf16* __restrict__ x2h) {
    int row = blockIdx.x, tid = threadIdx.x;
    const float* xr = h + (long)row * Hdim;
    float v[4], s = 0.f, sq = 0.f;
#pragma unroll
    for (int j = 0; j < 4; ++j) { v[j] = xr[tid * 4 + j]; s += v[j]; sq += v[j] * v[j]; }
    __shared__ float s1[256], s2[256];
    s1[tid] = s; s2[tid] = sq; __syncthreads();
    for (int st = 128; st > 0; st >>= 1) {
        if (tid < st) { s1[tid] += s1[tid + st]; s2[tid] += s2[tid + st]; }
        __syncthreads();
    }
    float mu = s1[0] * (1.f / Hdim);
    float var = s2[0] * (1.f / Hdim) - mu * mu;
    float rs = rsqrtf(var + 1e-5f);
#pragma unroll
    for (int j = 0; j < 4; ++j) {
        int c = tid * 4 + j;
        float y = ((v[j] - mu) * rs) * g[c] + b[c] + v[j];
        x2[(long)row * Hdim + c] = y;
        x2h[(long)row * Hdim + c] = (__bf16)y;
    }
}

// final LN over ymoe slot0+slot1
__global__ __launch_bounds__(256) void k_lnf(const float* __restrict__ ymoe,
                                             const float* __restrict__ g,
                                             const float* __restrict__ b,
                                             float* __restrict__ out) {
    int row = blockIdx.x, tid = threadIdx.x;
    const float* y0 = ymoe + (long)row * Hdim;
    const float* y1 = ymoe + (long)(NTOK + row) * Hdim;
    float v[4], s = 0.f, sq = 0.f;
#pragma unroll
    for (int j = 0; j < 4; ++j) {
        int c = tid * 4 + j;
        v[j] = y0[c] + y1[c]; s += v[j]; sq += v[j] * v[j];
    }
    __shared__ float s1[256], s2[256];
    s1[tid] = s; s2[tid] = sq; __syncthreads();
    for (int st = 128; st > 0; st >>= 1) {
        if (tid < st) { s1[tid] += s1[tid + st]; s2[tid] += s2[tid + st]; }
        __syncthreads();
    }
    float mu = s1[0] * (1.f / Hdim);
    float var = s2[0] * (1.f / Hdim) - mu * mu;
    float rs = rsqrtf(var + 1e-5f);
#pragma unroll
    for (int j = 0; j < 4; ++j) {
        int c = tid * 4 + j;
        out[(long)row * Hdim + c] = ((v[j] - mu) * rs) * g[c] + b[c];
    }
}

// ---------------- GEMM core: block 128(M) x 64(N), wave tile 32x32 ----------------
// waves: wm = wid>>1 (0..3) rows wm*32..+31 ; wn = wid&1 cols wn*32..+31
// Per K-chunk: 2 A-frags + 2 B-frags (8 ds_load_b128) -> 4 independent WMMAs.

// h1 = gelu(xh @ w1 + b1), bf16 out. grid(MLPH/64, NTOK/128)
__global__ __launch_bounds__(256) void k_mlp1(const __bf16* __restrict__ xh,
                                              const float* __restrict__ w1,
                                              const float* __restrict__ b1,
                                              __bf16* __restrict__ h1) {
    __shared__ __align__(16) __bf16 As[128 * LDT];
    __shared__ __align__(16) __bf16 Bs[64 * LDT];
    int tid = threadIdx.x, lane = tid & 31, wid = tid >> 5;
    int wm = wid >> 1, wn = wid & 1;
    int bn0 = blockIdx.x * 64, bm0 = blockIdx.y * 128;
    v8f a00 = {}, a01 = {}, a10 = {}, a11 = {};
    for (int kt = 0; kt < Hdim / 32; ++kt) {
        stage_a(xh + (long)bm0 * Hdim + kt * 32, Hdim, As, tid);
        stage_b(w1 + (long)(kt * 32) * MLPH + bn0, MLPH, Bs, tid);
        if (kt + 1 < Hdim / 32)
            __builtin_prefetch(w1 + (long)((kt + 1) * 32 + (tid >> 6) * 8) * MLPH + bn0 + (tid & 63), 0, 0);
        __syncthreads();
        v16bf fa0 = frag_ld(As, wm * 32 + (lane & 15), lane);
        v16bf fa1 = frag_ld(As, wm * 32 + 16 + (lane & 15), lane);
        v16bf fb0 = frag_ld(Bs, wn * 32 + (lane & 15), lane);
        v16bf fb1 = frag_ld(Bs, wn * 32 + 16 + (lane & 15), lane);
        a00 = wmma_bf16(fa0, fb0, a00);
        a01 = wmma_bf16(fa0, fb1, a01);
        a10 = wmma_bf16(fa1, fb0, a10);
        a11 = wmma_bf16(fa1, fb1, a11);
        __syncthreads();
    }
    int col0 = bn0 + wn * 32 + (lane & 15);
    int row0 = bm0 + wm * 32 + ((lane < 16) ? 0 : 8);
    float bb0 = b1[col0], bb1 = b1[col0 + 16];
#pragma unroll
    for (int r = 0; r < 8; ++r) {
        long ra = (long)(row0 + r) * MLPH;
        long rb = (long)(row0 + 16 + r) * MLPH;
        h1[ra + col0]      = (__bf16)gelu_f(a00[r] + bb0);
        h1[ra + col0 + 16] = (__bf16)gelu_f(a01[r] + bb1);
        h1[rb + col0]      = (__bf16)gelu_f(a10[r] + bb0);
        h1[rb + col0 + 16] = (__bf16)gelu_f(a11[r] + bb1);
    }
}

// hres = h1 @ w2 + b2 + x (residual). grid(Hdim/64, NTOK/128)
__global__ __launch_bounds__(256) void k_mlp2(const __bf16* __restrict__ h1,
                                              const float* __restrict__ w2,
                                              const float* __restrict__ b2,
                                              const float* __restrict__ xres,
                                              float* __restrict__ hres) {
    __shared__ __align__(16) __bf16 As[128 * LDT];
    __shared__ __align__(16) __bf16 Bs[64 * LDT];
    int tid = threadIdx.x, lane = tid & 31, wid = tid >> 5;
    int wm = wid >> 1, wn = wid & 1;
    int bn0 = blockIdx.x * 64, bm0 = blockIdx.y * 128;
    v8f a00 = {}, a01 = {}, a10 = {}, a11 = {};
    for (int kt = 0; kt < MLPH / 32; ++kt) {
        stage_a(h1 + (long)bm0 * MLPH + kt * 32, MLPH, As, tid);
        stage_b(w2 + (long)(kt * 32) * Hdim + bn0, Hdim, Bs, tid);
        if (kt + 1 < MLPH / 32)
            __builtin_prefetch(w2 + (long)((kt + 1) * 32 + (tid >> 6) * 8) * Hdim + bn0 + (tid & 63), 0, 0);
        __syncthreads();
        v16bf fa0 = frag_ld(As, wm * 32 + (lane & 15), lane);
        v16bf fa1 = frag_ld(As, wm * 32 + 16 + (lane & 15), lane);
        v16bf fb0 = frag_ld(Bs, wn * 32 + (lane & 15), lane);
        v16bf fb1 = frag_ld(Bs, wn * 32 + 16 + (lane & 15), lane);
        a00 = wmma_bf16(fa0, fb0, a00);
        a01 = wmma_bf16(fa0, fb1, a01);
        a10 = wmma_bf16(fa1, fb0, a10);
        a11 = wmma_bf16(fa1, fb1, a11);
        __syncthreads();
    }
    int col0 = bn0 + wn * 32 + (lane & 15);
    int row0 = bm0 + wm * 32 + ((lane < 16) ? 0 : 8);
    float bb0 = b2[col0], bb1 = b2[col0 + 16];
#pragma unroll
    for (int r = 0; r < 8; ++r) {
        long oa = (long)(row0 + r) * Hdim + col0;
        long ob = (long)(row0 + 16 + r) * Hdim + col0;
        hres[oa]      = a00[r] + bb0 + xres[oa];
        hres[oa + 16] = a01[r] + bb1 + xres[oa + 16];
        hres[ob]      = a10[r] + bb0 + xres[ob];
        hres[ob + 16] = a11[r] + bb1 + xres[ob + 16];
    }
}

// ---------------- Router + dispatch ----------------
__global__ void k_zero(int* counts) { if (threadIdx.x < NE) counts[threadIdx.x] = 0; }

__global__ __launch_bounds__(256) void k_router(const float* __restrict__ x2,
                                                const float* __restrict__ rw,
                                                int* __restrict__ counts,
                                                int* __restrict__ btok,
                                                int* __restrict__ bslot,
                                                float* __restrict__ bwgt) {
    int tok = blockIdx.x * 8 + (threadIdx.x >> 5);
    int lane = threadIdx.x & 31;
    const float* xr = x2 + (long)tok * Hdim;
    float acc[NE];
#pragma unroll
    for (int e = 0; e < NE; ++e) acc[e] = 0.f;
    for (int i = lane; i < Hdim; i += 32) {
        float xv = xr[i];
        const float* w = rw + (long)i * NE;
#pragma unroll
        for (int e = 0; e < NE; ++e) acc[e] += xv * w[e];
    }
#pragma unroll
    for (int e = 0; e < NE; ++e)
        for (int m = 16; m >= 1; m >>= 1) acc[e] += __shfl_xor(acc[e], m, 32);
    if (lane == 0) {
        float mx = acc[0];
        for (int e = 1; e < NE; ++e) mx = fmaxf(mx, acc[e]);
        float p[NE], sum = 0.f;
        for (int e = 0; e < NE; ++e) { p[e] = expf(acc[e] - mx); sum += p[e]; }
        float inv = 1.f / sum;
        for (int e = 0; e < NE; ++e) p[e] *= inv;
        int i0 = 0;
        for (int e = 1; e < NE; ++e) if (p[e] > p[i0]) i0 = e;
        int i1 = (i0 == 0) ? 1 : 0;
        for (int e = 0; e < NE; ++e) if (e != i0 && p[e] > p[i1]) i1 = e;
        float t = p[i0] + p[i1];
        float w0 = p[i0] / t, w1v = p[i1] / t;
        int p0 = atomicAdd(&counts[i0], 1);
        btok[i0 * NTOK + p0] = tok; bslot[i0 * NTOK + p0] = 0; bwgt[i0 * NTOK + p0] = w0;
        int p1 = atomicAdd(&counts[i1], 1);
        btok[i1 * NTOK + p1] = tok; bslot[i1 * NTOK + p1] = 1; bwgt[i1 * NTOK + p1] = w1v;
    }
}

__global__ void k_offs_aux(const int* __restrict__ counts, int* __restrict__ offs,
                           float* __restrict__ aux) {
    if (threadIdx.x == 0 && blockIdx.x == 0) {
        int o = 0; float s = 0.f;
        const float total = 2.f * NTOK;
        for (int e = 0; e < NE; ++e) {
            offs[e] = o; o += counts[e];
            float load = (float)counts[e] / total;
            s += load * logf(load + 1e-9f);
        }
        *aux = 0.01f * s;
    }
}

// ---------------- Grouped expert GEMMs ----------------
// hbuf[offs[e]+pos] = gelu(x2h[btok] @ ew1[e] + eb1[e]), bf16. grid(EXPH/64, NTOK/128, NE)
__global__ __launch_bounds__(256) void k_exp1(const __bf16* __restrict__ x2h,
                                              const float* __restrict__ ew1,
                                              const float* __restrict__ eb1,
                                              const int* __restrict__ counts,
                                              const int* __restrict__ offs,
                                              const int* __restrict__ btok,
                                              __bf16* __restrict__ hbuf) {
    int e = blockIdx.z;
    int cnt = counts[e];
    int bm0 = blockIdx.y * 128;
    if (bm0 >= cnt) return;
    __shared__ __align__(16) __bf16 As[128 * LDT];
    __shared__ __align__(16) __bf16 Bs[64 * LDT];
    __shared__ int rows[128];
    int tid = threadIdx.x, lane = tid & 31, wid = tid >> 5;
    int wm = wid >> 1, wn = wid & 1;
    if (tid < 128) {
        int pos = bm0 + tid; if (pos >= cnt) pos = cnt - 1;
        rows[tid] = btok[e * NTOK + pos];
    }
    __syncthreads();
    int bn0 = blockIdx.x * 64;
    const float* Bbase = ew1 + (long)e * Hdim * EXPH + bn0;
    v8f a00 = {}, a01 = {}, a10 = {}, a11 = {};
    for (int kt = 0; kt < Hdim / 32; ++kt) {
        stage_a_rows(x2h, Hdim, rows, kt * 32, As, tid);
        stage_b(Bbase + (long)(kt * 32) * EXPH, EXPH, Bs, tid);
        __syncthreads();
        v16bf fa0 = frag_ld(As, wm * 32 + (lane & 15), lane);
        v16bf fa1 = frag_ld(As, wm * 32 + 16 + (lane & 15), lane);
        v16bf fb0 = frag_ld(Bs, wn * 32 + (lane & 15), lane);
        v16bf fb1 = frag_ld(Bs, wn * 32 + 16 + (lane & 15), lane);
        a00 = wmma_bf16(fa0, fb0, a00);
        a01 = wmma_bf16(fa0, fb1, a01);
        a10 = wmma_bf16(fa1, fb0, a10);
        a11 = wmma_bf16(fa1, fb1, a11);
        __syncthreads();
    }
    int col0 = bn0 + wn * 32 + (lane & 15);
    int rloc = wm * 32 + ((lane < 16) ? 0 : 8);
    float bb0 = eb1[e * EXPH + col0], bb1 = eb1[e * EXPH + col0 + 16];
    int ob = offs[e];
#pragma unroll
    for (int r = 0; r < 8; ++r) {
        int posA = bm0 + rloc + r;
        int posB = posA + 16;
        if (posA < cnt) {
            hbuf[(long)(ob + posA) * EXPH + col0]      = (__bf16)gelu_f(a00[r] + bb0);
            hbuf[(long)(ob + posA) * EXPH + col0 + 16] = (__bf16)gelu_f(a01[r] + bb1);
        }
        if (posB < cnt) {
            hbuf[(long)(ob + posB) * EXPH + col0]      = (__bf16)gelu_f(a10[r] + bb0);
            hbuf[(long)(ob + posB) * EXPH + col0 + 16] = (__bf16)gelu_f(a11[r] + bb1);
        }
    }
}

// ymoe[slot][tok] = (hbuf_rows @ ew2[e] + eb2[e]) * wgt. grid(Hdim/64, NTOK/128, NE)
__global__ __launch_bounds__(256) void k_exp2(const __bf16* __restrict__ hbuf,
                                              const float* __restrict__ ew2,
                                              const float* __restrict__ eb2,
                                              const int* __restrict__ counts,
                                              const int* __restrict__ offs,
                                              const int* __restrict__ btok,
                                              const int* __restrict__ bslot,
                                              const float* __restrict__ bwgt,
                                              float* __restrict__ ymoe) {
    int e = blockIdx.z;
    int cnt = counts[e];
    int bm0 = blockIdx.y * 128;
    if (bm0 >= cnt) return;
    __shared__ __align__(16) __bf16 As[128 * LDT];
    __shared__ __align__(16) __bf16 Bs[64 * LDT];
    __shared__ int rows[128];
    int tid = threadIdx.x, lane = tid & 31, wid = tid >> 5;
    int wm = wid >> 1, wn = wid & 1;
    int ob = offs[e];
    if (tid < 128) {
        int pos = bm0 + tid; if (pos >= cnt) pos = cnt - 1;
        rows[tid] = ob + pos;
    }
    __syncthreads();
    int bn0 = blockIdx.x * 64;
    const float* Bbase = ew2 + (long)e * EXPH * Hdim + bn0;
    v8f a00 = {}, a01 = {}, a10 = {}, a11 = {};
    for (int kt = 0; kt < EXPH / 32; ++kt) {
        stage_a_rows(hbuf, EXPH, rows, kt * 32, As, tid);
        stage_b(Bbase + (long)(kt * 32) * Hdim, Hdim, Bs, tid);
        __syncthreads();
        v16bf fa0 = frag_ld(As, wm * 32 + (lane & 15), lane);
        v16bf fa1 = frag_ld(As, wm * 32 + 16 + (lane & 15), lane);
        v16bf fb0 = frag_ld(Bs, wn * 32 + (lane & 15), lane);
        v16bf fb1 = frag_ld(Bs, wn * 32 + 16 + (lane & 15), lane);
        a00 = wmma_bf16(fa0, fb0, a00);
        a01 = wmma_bf16(fa0, fb1, a01);
        a10 = wmma_bf16(fa1, fb0, a10);
        a11 = wmma_bf16(fa1, fb1, a11);
        __syncthreads();
    }
    int col0 = bn0 + wn * 32 + (lane & 15);
    int rloc = wm * 32 + ((lane < 16) ? 0 : 8);
    float bb0 = eb2[e * Hdim + col0], bb1 = eb2[e * Hdim + col0 + 16];
#pragma unroll
    for (int r = 0; r < 8; ++r) {
#pragma unroll
        for (int half = 0; half < 2; ++half) {
            int pos = bm0 + rloc + r + half * 16;
            if (pos < cnt) {
                int idx = e * NTOK + pos;
                int tok = btok[idx], slot = bslot[idx];
                float w = bwgt[idx];
                long o0 = ((long)slot * NTOK + tok) * Hdim + col0;
                float v0 = (half == 0) ? a00[r] : a10[r];
                float v1 = (half == 0) ? a01[r] : a11[r];
                ymoe[o0]      = (v0 + bb0) * w;
                ymoe[o0 + 16] = (v1 + bb1) * w;
            }
        }
    }
}

// ---------------- launcher ----------------
extern "C" void kernel_launch(void* const* d_in, const int* in_sizes, int n_in,
                              void* d_out, int out_size, void* d_ws, size_t ws_size,
                              hipStream_t stream) {
    (void)in_sizes; (void)n_in; (void)out_size; (void)ws_size;
    const float* x        = (const float*)d_in[0];
    const float* ln1_g    = (const float*)d_in[1];
    const float* ln1_b    = (const float*)d_in[2];
    const float* ln2_g    = (const float*)d_in[3];
    const float* ln2_b    = (const float*)d_in[4];
    const float* w1       = (const float*)d_in[5];
    const float* b1       = (const float*)d_in[6];
    const float* w2       = (const float*)d_in[7];
    const float* b2       = (const float*)d_in[8];
    const float* router_w = (const float*)d_in[9];
    const float* ew1      = (const float*)d_in[10];
    const float* eb1      = (const float*)d_in[11];
    const float* ew2      = (const float*)d_in[12];
    const float* eb2      = (const float*)d_in[13];
    const float* lnf_g    = (const float*)d_in[14];
    const float* lnf_b    = (const float*)d_in[15];
    float* out = (float*)d_out;

    char* ws = (char*)d_ws;
    size_t off = 0;
    auto alloc = [&](size_t bytes) -> char* {
        char* p = ws + off;
        off = (off + bytes + 255) & ~(size_t)255;
        return p;
    };
    __bf16* xh   = (__bf16*)alloc((size_t)NTOK * Hdim * 2);   // LN1 output (bf16)
    __bf16* h1   = (__bf16*)alloc((size_t)NTOK * MLPH * 2);   // gelu(x@w1+b1)
    float*  hres = (float*) alloc((size_t)NTOK * Hdim * 4);   // mlp out + residual
    float*  x2   = (float*) alloc((size_t)NTOK * Hdim * 4);   // LN2+res (f32)
    __bf16* x2h  = (__bf16*)alloc((size_t)NTOK * Hdim * 2);   // LN2+res (bf16)
    int*    counts = (int*) alloc(NE * 4);
    int*    offs   = (int*) alloc(NE * 4);
    int*    btok   = (int*) alloc((size_t)NE * NTOK * 4);
    int*    bslot  = (int*) alloc((size_t)NE * NTOK * 4);
    float*  bwgt   = (float*)alloc((size_t)NE * NTOK * 4);
    __bf16* hbuf = (__bf16*)alloc((size_t)2 * NTOK * EXPH * 2); // packed expert hidden
    float*  ymoe = (float*) alloc((size_t)2 * NTOK * Hdim * 4); // per-slot outputs

    // ResMLP block
    k_ln1 <<<NTOK, 256, 0, stream>>>(x, ln1_g, ln1_b, xh);
    k_mlp1<<<dim3(MLPH / 64, NTOK / 128), 256, 0, stream>>>(xh, w1, b1, h1);
    k_mlp2<<<dim3(Hdim / 64, NTOK / 128), 256, 0, stream>>>(h1, w2, b2, x, hres);
    k_ln2 <<<NTOK, 256, 0, stream>>>(hres, ln2_g, ln2_b, x2, x2h);

    // Router + dispatch
    k_zero  <<<1, 32, 0, stream>>>(counts);
    k_router<<<NTOK / 8, 256, 0, stream>>>(x2, router_w, counts, btok, bslot, bwgt);
    k_offs_aux<<<1, 1, 0, stream>>>(counts, offs, out + (size_t)NTOK * Hdim);

    // Grouped expert FFN (top-2 only)
    k_exp1<<<dim3(EXPH / 64, NTOK / 128, NE), 256, 0, stream>>>(x2h, ew1, eb1, counts, offs, btok, hbuf);
    k_exp2<<<dim3(Hdim / 64, NTOK / 128, NE), 256, 0, stream>>>(hbuf, ew2, eb2, counts, offs, btok, bslot, bwgt, ymoe);

    // Final LN
    k_lnf<<<NTOK, 256, 0, stream>>>(ymoe, lnf_g, lnf_b, out);
}
